// HeteroGraphSAGE_85555748536949
// MI455X (gfx1250) — matrix operand order repaired and checked by reading, not compiled, not run on hardware
//
#include <hip/hip_runtime.h>
#include <hip/hip_bf16.h>
#include <math.h>

typedef __attribute__((ext_vector_type(16))) _Float16 v16h;
typedef __attribute__((ext_vector_type(8)))  _Float16 v8h;
typedef __attribute__((ext_vector_type(8)))  float    v8f;

#define N_USER 100000
#define N_ITEM 100000
#define E_EDGES 800000
#define D_IN 128
#define HDIM 256
#define D_OUT 256

// ---------------------------------------------------------------------------
// Fragment layouts per CDNA5 ISA 7.12.2 (wave32), f16 16x16x32 WMMA:
//   A 16x32: lane r(0..15) row M=r; half=lane>>4; elems 0..7 -> K=half*8+e,
//            elems 8..15 -> K=16+half*8+(e-8)  => two contiguous 16B runs
//   B 32x16: lane col N=r; same K order as A (elems 0..7 -> K=half*8+e, 8..15 -> K=16+half*8+(e-8))
//            => in col-major Bt: two contiguous 16B runs, as for A
//   D 16x16: VGPR f -> M=f+half*8, N=r
// ---------------------------------------------------------------------------
template <int K>
__device__ __forceinline__ v8f wmma_chain(const _Float16* __restrict__ Arow,
                                          const _Float16* __restrict__ Bcol,
                                          v8f acc)
{
#pragma unroll
  for (int k0 = 0; k0 < K; k0 += 32) {
    if (k0 + 32 < K) {
      __builtin_prefetch(Arow + k0 + 32, 0, 3);
      __builtin_prefetch(Bcol + k0 + 32, 0, 3);
    }
    const v8h a0 = *(const v8h*)(Arow + k0);       // K = k0 + half*8 + 0..7
    const v8h a1 = *(const v8h*)(Arow + k0 + 16);  // K = k0 + 16 + half*8 + 0..7
    const v8h b0 = *(const v8h*)(Bcol + k0);       // K = k0 + half*8 + 0..7
    const v8h b1 = *(const v8h*)(Bcol + k0 + 16);  // K = k0 + 16 + half*8 + 0..7
    const v16h a = __builtin_shufflevector(a0, a1, 0, 1, 2, 3, 4, 5, 6, 7,
                                           8, 9, 10, 11, 12, 13, 14, 15);
    const v16h b = __builtin_shufflevector(b0, b1, 0, 1, 2, 3, 4, 5, 6, 7,
                                           8, 9, 10, 11, 12, 13, 14, 15);
    acc = __builtin_amdgcn_wmma_f32_16x16x32_f16(
        false, a, false, b, (short)0, acc, false, false);
  }
  return acc;
}

__device__ __forceinline__ float gelu_exact(float v) {
  return 0.5f * v * (1.0f + erff(v * 0.70710678118654752f));
}

// C[M,N] = act(A[M,K] @ B + bias)   (A f16 row-major, Bt f16 col-major)
template <int K, bool GELU>
__global__ __launch_bounds__(128) void gemm_wmma_f16(
    const _Float16* __restrict__ A, const _Float16* __restrict__ Bt,
    const float* __restrict__ bias, _Float16* __restrict__ C, int N)
{
  const int wave = threadIdx.x >> 5;
  const int lane = threadIdx.x & 31;
  const int half = lane >> 4;
  const int r    = lane & 15;
  const int m0   = blockIdx.x * 16;
  const int col  = (blockIdx.y * 4 + wave) * 16 + r;

  const _Float16* Arow = A  + (size_t)(m0 + r) * K + half * 8;
  const _Float16* Bcol = Bt + (size_t)col * K + half * 8;

  v8f acc = {};
  acc = wmma_chain<K>(Arow, Bcol, acc);

  const float bv = bias[col];
#pragma unroll
  for (int f = 0; f < 8; ++f) {
    const size_t idx = (size_t)(m0 + half * 8 + f) * N + col;
    float v = acc[f] + bv;
    if (GELU) v = gelu_exact(v);
    C[idx] = (_Float16)v;
  }
}

// Fused dual product: C = gelu(A1 @ B1 + A2 @ B2 + bias), K = HDIM = 256.
// Two unrolled WMMA chains into one f32 accumulator; single store pass.
__global__ __launch_bounds__(128) void gemm2_wmma_f16(
    const _Float16* __restrict__ A1, const _Float16* __restrict__ Bt1,
    const _Float16* __restrict__ A2, const _Float16* __restrict__ Bt2,
    const float* __restrict__ bias, _Float16* __restrict__ C, int N)
{
  constexpr int K = HDIM;
  const int wave = threadIdx.x >> 5;
  const int lane = threadIdx.x & 31;
  const int half = lane >> 4;
  const int r    = lane & 15;
  const int m0   = blockIdx.x * 16;
  const int col  = (blockIdx.y * 4 + wave) * 16 + r;

  const size_t aoff = (size_t)(m0 + r) * K + half * 8;
  const size_t boff = (size_t)col * K + half * 8;

  v8f acc = {};
  acc = wmma_chain<K>(A1 + aoff, Bt1 + boff, acc);
  acc = wmma_chain<K>(A2 + aoff, Bt2 + boff, acc);

  const float bv = bias[col];
#pragma unroll
  for (int f = 0; f < 8; ++f) {
    const size_t idx = (size_t)(m0 + half * 8 + f) * N + col;
    C[idx] = (_Float16)gelu_exact(acc[f] + bv);
  }
}

// Convert fp32 [K,N] weight to f16 col-major [N,K]
__global__ __launch_bounds__(256) void w_to_f16t(
    const float* __restrict__ W, _Float16* __restrict__ Wt, int K, int N)
{
  const int i = blockIdx.x * 256 + threadIdx.x;
  if (i >= K * N) return;
  const int k = i / N, n = i - k * N;
  Wt[(size_t)n * K + k] = (_Float16)W[i];
}

__global__ __launch_bounds__(256) void f32_to_f16(
    const float* __restrict__ in, _Float16* __restrict__ out, long n)
{
  const long i = (long)blockIdx.x * 256 + threadIdx.x;
  if (i < n) out[i] = (_Float16)in[i];
}

// ---------------------------------------------------------------------------
// Edge scatter: agg[dst] += h[src] (f16 rows, L2-resident); cnt[dst] += 1.
// One wave per edge; lane loads 8 contiguous f16 (global_load_b128), then
// 8 global f32 atomic adds resolving at L2.
// ---------------------------------------------------------------------------
__global__ __launch_bounds__(256) void scatter_add(
    const _Float16* __restrict__ h, const int* __restrict__ src,
    const int* __restrict__ dst, float* __restrict__ agg,
    float* __restrict__ cnt, int nEdges)
{
  const int gid  = blockIdx.x * blockDim.x + threadIdx.x;
  const int edge = gid >> 5;
  const int lane = gid & 31;
  if (edge >= nEdges) return;

  const int s = src[edge];
  const int d = dst[edge];
  const v8h hv = *(const v8h*)(h + (size_t)s * HDIM + lane * 8);
  float* ad = agg + (size_t)d * HDIM + lane * 8;
#pragma unroll
  for (int j = 0; j < 8; ++j) atomicAdd(&ad[j], (float)hv[j]);
  if (lane == 0) atomicAdd(&cnt[d], 1.0f);
}

// agg16[i][:] = f16( agg32[i][:] / max(cnt[i],1) )
__global__ __launch_bounds__(256) void mean_to_f16(
    const float* __restrict__ agg, const float* __restrict__ cnt,
    _Float16* __restrict__ out, int nRows)
{
  const size_t i = (size_t)blockIdx.x * 256 + threadIdx.x;
  if (i >= (size_t)nRows * HDIM) return;
  out[i] = (_Float16)(agg[i] / fmaxf(cnt[i >> 8], 1.0f));
}

// ---------------------------------------------------------------------------
// LayerNorm over 256 f16 features -> fp32 out. One wave per row; one b128
// load per lane; wave32 __shfl_xor butterfly reductions.
// ---------------------------------------------------------------------------
__global__ __launch_bounds__(256) void ln_row(
    const _Float16* __restrict__ X, const float* __restrict__ g,
    const float* __restrict__ b, float* __restrict__ out, int nRows)
{
  const int gid  = blockIdx.x * blockDim.x + threadIdx.x;
  const int row  = gid >> 5;
  const int lane = gid & 31;
  if (row >= nRows) return;

  const v8h xv = *(const v8h*)(X + (size_t)row * D_OUT + lane * 8);
  float v[8];
  float s = 0.0f;
#pragma unroll
  for (int j = 0; j < 8; ++j) { v[j] = (float)xv[j]; s += v[j]; }
#pragma unroll
  for (int off = 16; off > 0; off >>= 1) s += __shfl_xor(s, off, 32);
  const float mu = s * (1.0f / 256.0f);

  float var = 0.0f;
#pragma unroll
  for (int j = 0; j < 8; ++j) { const float d = v[j] - mu; var += d * d; }
#pragma unroll
  for (int off = 16; off > 0; off >>= 1) var += __shfl_xor(var, off, 32);
  var *= (1.0f / 256.0f);

  const float rs = rsqrtf(var + 1e-5f);
  float* o = out + (size_t)row * D_OUT;
#pragma unroll
  for (int j = 0; j < 8; ++j) {
    const int c = lane * 8 + j;
    o[c] = (v[j] - mu) * rs * g[c] + b[c];
  }
}

// ---------------------------------------------------------------------------
extern "C" void kernel_launch(void* const* d_in, const int* in_sizes, int n_in,
                              void* d_out, int out_size, void* d_ws, size_t ws_size,
                              hipStream_t stream) {
  const float* x_user = (const float*)d_in[0];
  const float* x_item = (const float*)d_in[1];
  const int* ui_src = (const int*)d_in[2];
  const int* ui_dst = (const int*)d_in[3];
  const int* iu_src = (const int*)d_in[4];
  const int* iu_dst = (const int*)d_in[5];
  const float* lin_user_W = (const float*)d_in[6];
  const float* lin_user_b = (const float*)d_in[7];
  const float* lin_item_W = (const float*)d_in[8];
  const float* lin_item_b = (const float*)d_in[9];
  const float* cW[2][2][3];   // [layer][et: 0=ui 1=iu][0=Wl 1=bl 2=Wr]
  for (int l = 0; l < 2; ++l)
    for (int t = 0; t < 2; ++t)
      for (int p = 0; p < 3; ++p)
        cW[l][t][p] = (const float*)d_in[10 + l * 6 + t * 3 + p];
  const float* out_user_W = (const float*)d_in[22];
  const float* out_user_b = (const float*)d_in[23];
  const float* out_item_W = (const float*)d_in[24];
  const float* out_item_b = (const float*)d_in[25];
  const float* ln_user_g  = (const float*)d_in[26];
  const float* ln_user_b2 = (const float*)d_in[27];
  const float* ln_item_g  = (const float*)d_in[28];
  const float* ln_item_b2 = (const float*)d_in[29];

  float* out = (float*)d_out;   // [out_u | out_i], fp32

  // ---- workspace carve ----------------------------------------------------
  char* cur = (char*)d_ws;
  auto take = [&](size_t bytes) {
    void* p = (void*)cur;
    cur += (bytes + 255) & ~(size_t)255;
    return p;
  };
  const size_t NH = (size_t)N_USER * HDIM;           // 25.6M elems
  _Float16* xu16 = (_Float16*)take((size_t)N_USER * D_IN * 2);
  _Float16* xi16 = (_Float16*)take((size_t)N_ITEM * D_IN * 2);
  _Float16* huA  = (_Float16*)take(NH * 2);
  _Float16* hiA  = (_Float16*)take(NH * 2);
  _Float16* huB  = (_Float16*)take(NH * 2);
  _Float16* hiB  = (_Float16*)take(NH * 2);
  float*    agU  = (float*)take(NH * 4);
  float*    agI  = (float*)take(NH * 4);
  _Float16* ag16U = (_Float16*)take(NH * 2);
  _Float16* ag16I = (_Float16*)take(NH * 2);
  float*    cntU = (float*)take((size_t)N_USER * 4);
  float*    cntI = (float*)take((size_t)N_ITEM * 4);
  // f16 transposed weights
  _Float16* wLinU = (_Float16*)take((size_t)D_IN * HDIM * 2);
  _Float16* wLinI = (_Float16*)take((size_t)D_IN * HDIM * 2);
  _Float16* wC[2][2][2];      // [layer][et][0=Wl^T 1=Wr^T]
  for (int l = 0; l < 2; ++l)
    for (int t = 0; t < 2; ++t)
      for (int p = 0; p < 2; ++p)
        wC[l][t][p] = (_Float16*)take((size_t)HDIM * HDIM * 2);
  _Float16* wOutU = (_Float16*)take((size_t)HDIM * D_OUT * 2);
  _Float16* wOutI = (_Float16*)take((size_t)HDIM * D_OUT * 2);

  // ---- one-time weight convert+transpose & input down-convert -------------
  const int tp128 = (D_IN * HDIM + 255) / 256;
  const int tp256 = (HDIM * HDIM + 255) / 256;
  w_to_f16t<<<tp128, 256, 0, stream>>>(lin_user_W, wLinU, D_IN, HDIM);
  w_to_f16t<<<tp128, 256, 0, stream>>>(lin_item_W, wLinI, D_IN, HDIM);
  for (int l = 0; l < 2; ++l)
    for (int t = 0; t < 2; ++t) {
      w_to_f16t<<<tp256, 256, 0, stream>>>(cW[l][t][0], wC[l][t][0], HDIM, HDIM);
      w_to_f16t<<<tp256, 256, 0, stream>>>(cW[l][t][2], wC[l][t][1], HDIM, HDIM);
    }
  w_to_f16t<<<tp256, 256, 0, stream>>>(out_user_W, wOutU, HDIM, D_OUT);
  w_to_f16t<<<tp256, 256, 0, stream>>>(out_item_W, wOutI, HDIM, D_OUT);

  const long nx = (long)N_USER * D_IN;
  f32_to_f16<<<(int)((nx + 255) / 256), 256, 0, stream>>>(x_user, xu16, nx);
  f32_to_f16<<<(int)((nx + 255) / 256), 256, 0, stream>>>(x_item, xi16, nx);

  const dim3 gemmBlk(128);
  const dim3 gemmGrid(N_USER / 16, HDIM / 64);       // 6250 x 4 (M divisible by 16)
  const int scatBlocks = (E_EDGES * 32 + 255) / 256; // one wave per edge
  const int lnBlocks   = (N_USER * 32 + 255) / 256;
  const int meanBlocks = (int)((NH + 255) / 256);

  // ---- input projection + GELU -------------------------------------------
  gemm_wmma_f16<D_IN, true><<<gemmGrid, gemmBlk, 0, stream>>>(
      xu16, wLinU, lin_user_b, huA, HDIM);
  gemm_wmma_f16<D_IN, true><<<gemmGrid, gemmBlk, 0, stream>>>(
      xi16, wLinI, lin_item_b, hiA, HDIM);

  // ---- 2 hetero SAGE layers ----------------------------------------------
  for (int l = 0; l < 2; ++l) {
    hipMemsetAsync(agI, 0, NH * sizeof(float), stream);
    hipMemsetAsync(agU, 0, NH * sizeof(float), stream);
    hipMemsetAsync(cntI, 0, N_ITEM * sizeof(float), stream);
    hipMemsetAsync(cntU, 0, N_USER * sizeof(float), stream);

    scatter_add<<<scatBlocks, 256, 0, stream>>>(huA, ui_src, ui_dst, agI, cntI, E_EDGES);
    scatter_add<<<scatBlocks, 256, 0, stream>>>(hiA, iu_src, iu_dst, agU, cntU, E_EDGES);

    mean_to_f16<<<meanBlocks, 256, 0, stream>>>(agI, cntI, ag16I, N_ITEM);
    mean_to_f16<<<meanBlocks, 256, 0, stream>>>(agU, cntU, ag16U, N_USER);

    // h_i_new = gelu(agg_i @ Wl_ui + h_i @ Wr_ui + bl_ui)   (one fused pass)
    gemm2_wmma_f16<<<gemmGrid, gemmBlk, 0, stream>>>(
        ag16I, wC[l][0][0], hiA, wC[l][0][1], cW[l][0][1], hiB, HDIM);
    // h_u_new = gelu(agg_u @ Wl_iu + h_u @ Wr_iu + bl_iu)
    gemm2_wmma_f16<<<gemmGrid, gemmBlk, 0, stream>>>(
        ag16U, wC[l][1][0], huA, wC[l][1][1], cW[l][1][1], huB, HDIM);

    _Float16* t;
    t = huA; huA = huB; huB = t;
    t = hiA; hiA = hiB; hiB = t;
  }

  // ---- output heads + LayerNorm (reuse ag16 buffers as temporaries) -------
  gemm_wmma_f16<HDIM, false><<<gemmGrid, gemmBlk, 0, stream>>>(
      huA, wOutU, out_user_b, ag16U, D_OUT);
  ln_row<<<lnBlocks, 256, 0, stream>>>(ag16U, ln_user_g, ln_user_b2, out, N_USER);

  gemm_wmma_f16<HDIM, false><<<gemmGrid, gemmBlk, 0, stream>>>(
      hiA, wOutI, out_item_b, ag16I, D_OUT);
  ln_row<<<lnBlocks, 256, 0, stream>>>(ag16I, ln_item_g, ln_item_b2,
                                       out + (size_t)N_USER * D_OUT, N_ITEM);
}
